// NetConvEdgePool_69483980915378
// MI455X (gfx1250) — matrix operand order repaired and checked by reading, not compiled, not run on hardware
//
#include <hip/hip_runtime.h>
#include <hip/hip_bf16.h>

typedef _Float16 f16;
typedef _Float16 h8  __attribute__((ext_vector_type(8)));
typedef _Float16 h16 __attribute__((ext_vector_type(16)));
typedef float    v8f __attribute__((ext_vector_type(8)));

#define N_NODES 100000
#define N_EDGES 600000
#define EPRIME  (N_EDGES + N_NODES)   // 700000, incl. one self-loop per node
#define N_IN    16
#define HID     64
#define HEADS   8
#define N_CONVS 5

// ---------------- workspace layout (bytes, all offsets 256B aligned) --------
#define OFF_WP   0u                          // packed Wc fragments: 5*32768 f16 = 327,680 B
#define OFF_H    327680u                     // h   f32 [N,64]  = 25,600,000 B
#define OFF_HF   25927680u                   // h   f16 [N,64]  = 12,800,000 B
#define OFF_M    38727680u                   // m   f16 [N,512] = 102,400,000 B (fits in 192MB L2)
#define OFF_NUM  141127680u                  // num f32 [N,64]  = 25,600,000 B
#define OFF_DEN  166727680u                  // den f32 [N]     = 400,000 B
#define WS_NEED  167127680u

// ---------------------------------------------------------------------------
// zero num/den at start of every launch (harness poisons ws once; we must be
// deterministic across replays)
__global__ __launch_bounds__(256) void init_zero(float* num, float* den) {
    int t = blockIdx.x * 256 + threadIdx.x;
    if (t < N_NODES * HID) num[t] = 0.0f;
    if (t < N_NODES)       den[t] = 0.0f;
}

// h = relu(x @ W1 + b1); also keep an f16 copy for the WMMA GEMM
__global__ __launch_bounds__(256) void lin1_kernel(const float* __restrict__ x,
                                                   const float* __restrict__ W1,
                                                   const float* __restrict__ b1,
                                                   float* __restrict__ h,
                                                   f16* __restrict__ hf) {
    int t = blockIdx.x * 256 + threadIdx.x;          // over N*64
    int n = t >> 6, j = t & 63;
    float acc = b1[j];
    const float* xr = x + n * N_IN;
#pragma unroll
    for (int k = 0; k < N_IN; ++k) acc = fmaf(xr[k], W1[k * HID + j], acc);
    acc = fmaxf(acc, 0.0f);
    h[t]  = acc;
    hf[t] = (f16)acc;
}

// Repack Wc[i] (K=64 x N=512, f32) into per-lane-contiguous f16 B-fragments.
// B (32x16, 16-bit) layout: lanes 0-15 hold K={0..7,16..23}, lanes
// 16-31 hold K={8..15,24..31}; column N = lane&15.  Packed stream per (layer, nTile,
// kTile): 32 lanes x 16 halves contiguous -> one aligned 32B load per lane.
__global__ __launch_bounds__(256) void pack_w(const float* __restrict__ Wc,
                                              f16* __restrict__ wp) {
    int t = blockIdx.x * 256 + threadIdx.x;          // over 5*32768
    if (t >= N_CONVS * 32768) return;
    int hIdx = t & 15;
    int lane = (t >> 4) & 31;
    int kt   = (t >> 9) & 1;
    int nt   = (t >> 10) & 31;
    int i    = t >> 15;
    int K = (hIdx & 8) * 2 + ((lane >> 4) ? 8 : 0) + (hIdx & 7);
    int k = kt * 32 + K;
    int n = nt * 16 + (lane & 15);
    wp[t] = (f16)Wc[((size_t)i * HID + k) * (HEADS * HID) + n];
}

// m = h @ Wc[i]  via v_wmma_f32_16x16x32_f16.  One block = one 16-row tile,
// 4 waves each covering 128 of the 512 output columns (8 N-tiles x 2 K-steps).
// Also zeroes den[] for the upcoming edge pass of this layer.
__global__ __launch_bounds__(128) void gemm_m(const f16* __restrict__ hf,
                                              const f16* __restrict__ wp,
                                              f16* __restrict__ m16,
                                              float* __restrict__ den) {
    int gtid = blockIdx.x * 128 + threadIdx.x;
    if (gtid < N_NODES) den[gtid] = 0.0f;

    int tile = blockIdx.x;                // 6250 row tiles
    int wave = threadIdx.x >> 5;
    int lane = threadIdx.x & 31;
    int lo   = lane & 15;
    int hi   = lane >> 4;                 // 0 or 1
    int rowBase = tile * 16;

    // A fragment (16x32 f16), ISA 7.12.2 layout:
    //   lanes 0-15 : M=lane,   K = {0..7, 16..23} (+32*kb)
    //   lanes 16-31: M=lane-16,K = {8..15,24..31} (+32*kb)
    h16 a[2];
    const f16* rowA = hf + (size_t)(rowBase + lo) * HID;
#pragma unroll
    for (int kb = 0; kb < 2; ++kb) {
        const f16* p = rowA + kb * 32 + (hi ? 8 : 0);
        h8 p0 = *(const h8*)(p);          // K = k0..k0+7
        h8 p1 = *(const h8*)(p + 16);     // K = k0+16..k0+23
        a[kb] = __builtin_shufflevector(p0, p1, 0, 1, 2, 3, 4, 5, 6, 7,
                                        8, 9, 10, 11, 12, 13, 14, 15);
    }

#pragma unroll
    for (int nb = 0; nb < 8; ++nb) {
        int nt = wave * 8 + nb;           // global 16-col tile 0..31
        v8f acc = {0.f, 0.f, 0.f, 0.f, 0.f, 0.f, 0.f, 0.f};
#pragma unroll
        for (int kb = 0; kb < 2; ++kb) {
            h16 b = *(const h16*)(wp + ((size_t)(nt * 2 + kb) * 32 + lane) * 16);
            acc = __builtin_amdgcn_wmma_f32_16x16x32_f16(
                false, a[kb], false, b, (short)0, acc, false, false);
        }
        // D layout: VGPR r -> row rowBase + r + 8*hi, col = nt*16 + lo
        int colBase = nt * 16;
#pragma unroll
        for (int r = 0; r < 8; ++r) {
            int row = rowBase + r + hi * 8;
            m16[(size_t)row * 512 + colBase + lo] = (f16)acc[r];
        }
    }
}

// One wave per edge: attention softmax over 8 heads, blend m[src] head rows,
// scatter-add into num/den with f32 global atomics (L2-resident).
__global__ __launch_bounds__(256) void edge_kernel(const int* __restrict__ ei,
                                                   const float* __restrict__ h,
                                                   const f16* __restrict__ m16,
                                                   const float* __restrict__ u,
                                                   const float* __restrict__ c,
                                                   float* __restrict__ num,
                                                   float* __restrict__ den) {
    int wv = blockIdx.x * 8 + (threadIdx.x >> 5);
    if (wv >= EPRIME) return;
    int lane = threadIdx.x & 31;

    int src, dst;
    if (wv < N_EDGES) {
        src = ei[wv];
        dst = ei[N_EDGES + wv];
        if (src == dst) return;           // w_edge == 0: contributes nothing
    } else {
        src = dst = wv - N_EDGES;         // self loop, w_edge == 1
    }

    float t0 = h[src * HID + lane]      - h[dst * HID + lane];
    float t1 = h[src * HID + lane + 32] - h[dst * HID + lane + 32];

    float logit[HEADS];
#pragma unroll
    for (int hh = 0; hh < HEADS; ++hh) {
        float p = fmaf(t0, u[lane * HEADS + hh],
                       t1 * u[(lane + 32) * HEADS + hh]);
#pragma unroll
        for (int m = 16; m >= 1; m >>= 1) p += __shfl_xor(p, m, 32);
        logit[hh] = p + c[hh];
    }
    float mx = logit[0];
#pragma unroll
    for (int hh = 1; hh < HEADS; ++hh) mx = fmaxf(mx, logit[hh]);
    float q[HEADS], s = 0.0f;
#pragma unroll
    for (int hh = 0; hh < HEADS; ++hh) { q[hh] = __expf(logit[hh] - mx); s += q[hh]; }
    float inv = 1.0f / s;

    const f16* mrow = m16 + (size_t)src * 512;
    float o0 = 0.0f, o1 = 0.0f;
#pragma unroll
    for (int hh = 0; hh < HEADS; ++hh) {
        float qq = q[hh] * inv;
        o0 = fmaf(qq, (float)mrow[hh * HID + lane],      o0);
        o1 = fmaf(qq, (float)mrow[hh * HID + lane + 32], o1);
    }
    unsafeAtomicAdd(&num[dst * HID + lane],      o0);
    unsafeAtomicAdd(&num[dst * HID + lane + 32], o1);
    if (lane == 0) unsafeAtomicAdd(&den[dst], 1.0f);
}

// h = relu(num/den + bc); re-zero num elementwise for next layer
__global__ __launch_bounds__(256) void finalize_kernel(float* __restrict__ num,
                                                       const float* __restrict__ den,
                                                       const float* __restrict__ bc,
                                                       float* __restrict__ h,
                                                       f16* __restrict__ hf) {
    int t = blockIdx.x * 256 + threadIdx.x;          // over N*64
    int n = t >> 6, j = t & 63;
    float v = num[t] / den[n] + bc[j];
    v = fmaxf(v, 0.0f);
    h[t]  = v;
    hf[t] = (f16)v;
    num[t] = 0.0f;
}

// out = relu(h @ W2 + b2) @ W3 + b3 ; one wave per node (lane = hidden/2 col)
__global__ __launch_bounds__(256) void head_kernel(const float* __restrict__ h,
                                                   const float* __restrict__ W2,
                                                   const float* __restrict__ b2,
                                                   const float* __restrict__ W3,
                                                   const float* __restrict__ b3,
                                                   float* __restrict__ out) {
    int n = blockIdx.x * 8 + (threadIdx.x >> 5);
    if (n >= N_NODES) return;
    int lane = threadIdx.x & 31;
    const float* hr = h + (size_t)n * HID;
    float acc = b2[lane];
#pragma unroll
    for (int k = 0; k < HID; ++k) acc = fmaf(hr[k], W2[k * 32 + lane], acc);
    acc = fmaxf(acc, 0.0f);
    float p0 = acc * W3[lane * 2 + 0];
    float p1 = acc * W3[lane * 2 + 1];
#pragma unroll
    for (int m = 16; m >= 1; m >>= 1) {
        p0 += __shfl_xor(p0, m, 32);
        p1 += __shfl_xor(p1, m, 32);
    }
    if (lane == 0) {
        out[n * 2 + 0] = p0 + b3[0];
        out[n * 2 + 1] = p1 + b3[1];
    }
}

extern "C" void kernel_launch(void* const* d_in, const int* in_sizes, int n_in,
                              void* d_out, int out_size, void* d_ws, size_t ws_size,
                              hipStream_t stream) {
    const float* x  = (const float*)d_in[0];
    const int*   ei = (const int*)  d_in[1];   // [2, E] int32 (jax x64 disabled)
    const float* W1 = (const float*)d_in[2];
    const float* b1 = (const float*)d_in[3];
    const float* Wc = (const float*)d_in[4];   // [5,64,512]
    const float* uc = (const float*)d_in[5];   // [5,64,8]
    const float* cc = (const float*)d_in[6];   // [5,8]
    const float* bc = (const float*)d_in[7];   // [5,64]
    const float* W2 = (const float*)d_in[8];
    const float* b2 = (const float*)d_in[9];
    const float* W3 = (const float*)d_in[10];
    const float* b3 = (const float*)d_in[11];
    float* out = (float*)d_out;

    char* ws = (char*)d_ws;
    f16*   wp  = (f16*)  (ws + OFF_WP);
    float* h   = (float*)(ws + OFF_H);
    f16*   hf  = (f16*)  (ws + OFF_HF);
    f16*   m16 = (f16*)  (ws + OFF_M);
    float* num = (float*)(ws + OFF_NUM);
    float* den = (float*)(ws + OFF_DEN);

    const int NE = N_NODES * HID;                        // 6,400,000

    init_zero<<<(NE + 255) / 256, 256, 0, stream>>>(num, den);
    lin1_kernel<<<NE / 256, 256, 0, stream>>>(x, W1, b1, h, hf);
    pack_w<<<(N_CONVS * 32768 + 255) / 256, 256, 0, stream>>>(Wc, wp);

    for (int i = 0; i < N_CONVS; ++i) {
        gemm_m<<<N_NODES / 16, 128, 0, stream>>>(hf, wp + (size_t)i * 32768, m16, den);
        edge_kernel<<<EPRIME / 8, 256, 0, stream>>>(ei, h, m16,
                                                    uc + (size_t)i * HID * HEADS,
                                                    cc + (size_t)i * HEADS, num, den);
        finalize_kernel<<<NE / 256, 256, 0, stream>>>(num, den, bc + (size_t)i * HID, h, hf);
    }

    head_kernel<<<(N_NODES + 7) / 8, 256, 0, stream>>>(h, W2, b2, W3, b3, out);
}
